// MultiHeadAttention_42554535969278
// MI455X (gfx1250) — compile-verified, matrix operand order checked by reading, not run on hardware
//
#include <hip/hip_runtime.h>
#include <hip/hip_bf16.h>

// ---------------------------------------------------------------------------
// MHA forward for MI455X (gfx1250), bf16 WMMA everywhere.
//   B=2, S=2048, D=1024, H=16, HD=64.  M = B*S = 4096.
// Pipeline:
//   1) f32->bf16 convert of activations; f32->bf16 transpose of weights (Wt[N][K])
//   2) QKV projections: v_wmma_f32_16x16x32_bf16; weight tiles staged into LDS
//      by the Tensor Data Mover (tensor_load_to_lds), double-buffered,
//      synchronized with s_wait_tensorcnt + workgroup barrier.
//   3) causal flash attention, online softmax, bf16 WMMA for QK^T and PV
//   4) output projection -> f32 d_out
// Workspace use: ~67 MB of d_ws.
// ---------------------------------------------------------------------------

typedef __bf16 bf16;
typedef __attribute__((ext_vector_type(16))) __bf16 v16bf;
typedef __attribute__((ext_vector_type(8)))  __bf16 v8bf;
typedef __attribute__((ext_vector_type(8)))  float  v8f;
typedef __attribute__((ext_vector_type(4)))  unsigned int u32x4;
typedef __attribute__((ext_vector_type(4)))  int i32x4;
typedef __attribute__((ext_vector_type(8)))  int i32x8;

static constexpr int Bn = 2, Sn = 2048, Dn = 1024, Hn = 16, HDn = 64;
static constexpr int Mn = Bn * Sn;   // 4096

union FragU { v16bf f; v8bf h[2]; };

static __device__ __forceinline__ v16bf make_frag(v8bf lo, v8bf hi) {
  FragU u; u.h[0] = lo; u.h[1] = hi; return u.f;
}

static __device__ __forceinline__ v8f wmma_bf16(v16bf a, v16bf b, v8f c) {
  // D = A(16x32) * B(32x16) + C, f32 accumulate
  return __builtin_amdgcn_wmma_f32_16x16x32_bf16(false, a, false, b, (short)0, c,
                                                 false, false);
}

static __device__ __forceinline__ v8f vzero8() {
  v8f z = {0.f, 0.f, 0.f, 0.f, 0.f, 0.f, 0.f, 0.f};
  return z;
}

// ------------------------- Tensor Data Mover (TDM) -------------------------
// DMA a 64(n-rows) x 32(k) bf16 tile of Wt[N][K] (row stride K=1024 elements)
// from global into LDS. D# packing per CDNA5 ISA:
//   g0: [1:0]=count=1, [63:32]=lds_addr, [120:64]=global_addr, [127:126]=type 2
//   g1: data_size=2B, tensor_dim0/1=1024, tile_dim0=32, tile_dim1=64,
//       tensor_dim0_stride=1024, everything else 0 (2-D, no pad/iterate).
static __device__ __forceinline__ void tdm_load_wt_tile(const bf16* gsrc,
                                                        bf16* ldst) {
  const unsigned long long ga = (unsigned long long)(uintptr_t)gsrc;
  const unsigned int lds = (unsigned int)(uintptr_t)ldst;  // LDS byte offset
  u32x4 g0;
  g0[0] = 1u;                                              // count = 1 (valid)
  g0[1] = lds;                                             // lds_addr
  g0[2] = (unsigned int)(ga & 0xffffffffull);              // global_addr lo
  g0[3] = ((unsigned int)(ga >> 32) & 0x01ffffffu) | (2u << 30);  // hi | type=2
  i32x8 g1;
  g1[0] = (int)(1u << 16);          // workgroup_mask=0, data_size=1 (2 bytes)
  g1[1] = (int)(1024u << 16);       // tensor_dim0[15:0]=1024 in [31:16]
  g1[2] = (int)(1024u << 16);       // dim0 hi=0, tensor_dim1[15:0]=1024
  g1[3] = (int)(32u << 16);         // dim1 hi=0, tile_dim0=32
  g1[4] = 64;                       // tile_dim1=64, tile_dim2=0
  g1[5] = 1024;                     // tensor_dim0_stride lo32 = 1024
  g1[6] = 0;                        // stride0 hi=0, stride1 lo=0
  g1[7] = 0;                        // stride1 hi=0
  i32x4 gz = {0, 0, 0, 0};
#if defined(__clang_major__) && (__clang_major__ >= 23)
  i32x8 gz8 = {0, 0, 0, 0, 0, 0, 0, 0};
  __builtin_amdgcn_tensor_load_to_lds(g0, g1, gz, gz, gz8, 0);
#else
  __builtin_amdgcn_tensor_load_to_lds(g0, g1, gz, gz, 0);
#endif
}

// --------------------------- convert / transpose ---------------------------

__global__ void k_cvt_f32_bf16(const float* __restrict__ in,
                               bf16* __restrict__ out, int n) {
  int i = blockIdx.x * blockDim.x + threadIdx.x;
  int stride = gridDim.x * blockDim.x;
  for (; i < n; i += stride) out[i] = (bf16)in[i];
}

// W[K=1024][N=1024] (row-major) -> Wt[N][K] bf16
__global__ void k_transpose_w(const float* __restrict__ W, bf16* __restrict__ Wt) {
  int n = blockIdx.x * 16 + threadIdx.x;
  int k = blockIdx.y * 16 + threadIdx.y;
  Wt[(size_t)n * Dn + k] = (bf16)W[(size_t)k * Dn + n];
}

// --------------------------------- GEMM ------------------------------------
// C[M,N] = A[M,K] * Wt[N,K]^T + bias. A block = 8 waves sharing one 64-column
// weight panel; each wave owns a 32x64 tile (256 rows x 64 cols per block).
// Wt k-slabs (64x32 bf16, 4 KB) are DMA'd into LDS by the TDM, double-buffered;
// A fragments are direct contiguous global b128 loads.
// MODE 0: bf16 out, [B,H,S,HD] layout (Q/K), scaled.
// MODE 1: bf16 out, [B,H,HD,S] layout (V transposed).
// MODE 2: f32 out, [M,N] row-major (final projection -> d_out).
template <int MODE>
__global__ __launch_bounds__(256) void k_gemm(const bf16* __restrict__ A,
                                              const bf16* __restrict__ Wt,
                                              const float* __restrict__ bias,
                                              bf16* __restrict__ outb,
                                              float* __restrict__ outf,
                                              float scale) {
  constexpr int N = Dn, K = Dn;
  __shared__ bf16 Wlds[2][64 * 32];   // double-buffered weight panel (2 x 4 KB)

  const int lane = threadIdx.x & 31;
  const int wave = threadIdx.x >> 5;
  const int nt = blockIdx.x & 15;     // 16 column panels of 64
  const int mg = blockIdx.x >> 4;     // 16 row groups of 256
  const int ln = lane & 15, kh = lane >> 4;
  const int m0 = (mg * 8 + wave) * 32;
  const int n0 = nt * 64;

  v8f acc[2][4];
#pragma unroll
  for (int im = 0; im < 2; ++im)
#pragma unroll
    for (int jn = 0; jn < 4; ++jn) acc[im][jn] = vzero8();

  // prologue: stage first weight slab via TDM (wave 0 issues the DMA)
  if (threadIdx.x < 32) {
    tdm_load_wt_tile(Wt + (size_t)n0 * K, &Wlds[0][0]);
    __builtin_amdgcn_s_wait_tensorcnt(0);
  }
  __syncthreads();

  int buf = 0;
  for (int k0 = 0; k0 < K; k0 += 32) {
    // kick off DMA of the next slab into the other buffer
    if ((k0 + 32 < K) && threadIdx.x < 32)
      tdm_load_wt_tile(Wt + (size_t)n0 * K + (k0 + 32), &Wlds[buf ^ 1][0]);

    v16bf af[2], bfr[4];
#pragma unroll
    for (int im = 0; im < 2; ++im) {
      const bf16* p = A + (size_t)(m0 + im * 16 + ln) * K + k0 + kh * 8;
      af[im] = make_frag(*(const v8bf*)p, *(const v8bf*)(p + 16));
    }
#pragma unroll
    for (int jn = 0; jn < 4; ++jn) {
      const bf16* p = &Wlds[buf][(jn * 16 + ln) * 32 + kh * 8];
      bfr[jn] = make_frag(*(const v8bf*)p, *(const v8bf*)(p + 16));
    }
#pragma unroll
    for (int im = 0; im < 2; ++im)
#pragma unroll
      for (int jn = 0; jn < 4; ++jn)
        acc[im][jn] = wmma_bf16(af[im], bfr[jn], acc[im][jn]);

    // publish the next slab: TDM completion, then block-wide visibility
    if (threadIdx.x < 32) __builtin_amdgcn_s_wait_tensorcnt(0);
    __syncthreads();
    buf ^= 1;
  }

  // epilogue: D tile layout = lane(ln) -> column, vgpr v -> row v + 8*kh
#pragma unroll
  for (int jn = 0; jn < 4; ++jn) {
    const int n = n0 + jn * 16 + ln;
    const float bn = bias[n];
#pragma unroll
    for (int im = 0; im < 2; ++im) {
#pragma unroll
      for (int v = 0; v < 8; ++v) {
        const int gm = m0 + im * 16 + v + kh * 8;
        const float val = (acc[im][jn][v] + bn) * scale;
        if (MODE == 2) {
          outf[(size_t)gm * N + n] = val;
        } else {
          const int b = gm >> 11, s = gm & (Sn - 1);
          const int h = n >> 6, hd = n & 63;
          size_t idx;
          if (MODE == 0)
            idx = (((size_t)(b * Hn + h)) * Sn + s) * HDn + hd;
          else
            idx = (((size_t)(b * Hn + h)) * HDn + hd) * Sn + s;
          outb[idx] = (bf16)val;
        }
      }
    }
  }
}

// ----------------------------- flash attention -----------------------------
// grid = B*H*(S/64) blocks, 128 threads (4 waves). Each wave: 16 query rows,
// streams causal keys in blocks of 64. Q pre-scaled by 1/sqrt(HD).
__global__ __launch_bounds__(128) void k_attn(const bf16* __restrict__ Q,   // [B,H,S,HD]
                                              const bf16* __restrict__ Kt,  // [B,H,S,HD]
                                              const bf16* __restrict__ Vt,  // [B,H,HD,S]
                                              bf16* __restrict__ O) {       // [B*S, D]
  __shared__ bf16 Plds[4][16 * 64];  // per-wave P tile (q x key), bf16

  const int lane = threadIdx.x & 31;
  const int wv   = threadIdx.x >> 5;      // 0..3
  const int ln = lane & 15, kh = lane >> 4;
  const int bh    = blockIdx.x >> 5;      // b*H + h   (S/64 == 32)
  const int qblk  = (blockIdx.x & 31) * 64;
  const int qbase = qblk + wv * 16;

  const bf16* qp = Q  + ((size_t)bh * Sn + qbase) * HDn;
  const bf16* kp = Kt + (size_t)bh * Sn * HDn;
  const bf16* vp = Vt + (size_t)bh * HDn * Sn;

  // Q fragments stay in registers for the whole key loop
  v16bf qf[2];
  {
    const bf16* p = qp + (size_t)ln * HDn + kh * 8;
    qf[0] = make_frag(*(const v8bf*)p,        *(const v8bf*)(p + 16));
    qf[1] = make_frag(*(const v8bf*)(p + 32), *(const v8bf*)(p + 48));
  }

  v8f o[4];
  float mi[8], li[8];
#pragma unroll
  for (int t = 0; t < 4; ++t) o[t] = vzero8();
#pragma unroll
  for (int v = 0; v < 8; ++v) { mi[v] = -1.0e30f; li[v] = 0.f; }

  const int kend = qbase + 16;  // exclusive causal key limit
  for (int kb = 0; kb < kend; kb += 64) {
    // prefetch next key/value blocks (speculative; OOB silently dropped)
    __builtin_prefetch(kp + (size_t)(kb + 64) * HDn, 0, 3);
    __builtin_prefetch(vp + (size_t)ln * Sn + kb + 64, 0, 3);

    // S = Q K^T : 4 key-tiles x (HD=64 -> 2 WMMAs)
    v8f st[4];
#pragma unroll
    for (int t = 0; t < 4; ++t) {
      st[t] = vzero8();
      const bf16* p = kp + (size_t)(kb + t * 16 + ln) * HDn + kh * 8;
      v16bf b0 = make_frag(*(const v8bf*)p,        *(const v8bf*)(p + 16));
      v16bf b1 = make_frag(*(const v8bf*)(p + 32), *(const v8bf*)(p + 48));
      st[t] = wmma_bf16(qf[0], b0, st[t]);
      st[t] = wmma_bf16(qf[1], b1, st[t]);
    }

    // causal mask on diagonal blocks (additive -1e9, matches reference)
    if (kb + 64 > qbase) {
#pragma unroll
      for (int t = 0; t < 4; ++t) {
        const int key = kb + t * 16 + ln;
#pragma unroll
        for (int v = 0; v < 8; ++v) {
          const int row = qbase + v + kh * 8;
          if (key > row) st[t][v] -= 1.0e9f;
        }
      }
    }

    // online softmax: row stats live per-lane for rows v + 8*kh
    float mnew[8], pscale[8];
#pragma unroll
    for (int v = 0; v < 8; ++v) {
      float mx = fmaxf(fmaxf(st[0][v], st[1][v]), fmaxf(st[2][v], st[3][v]));
      mx = fmaxf(mx, __shfl_xor(mx, 1));
      mx = fmaxf(mx, __shfl_xor(mx, 2));
      mx = fmaxf(mx, __shfl_xor(mx, 4));
      mx = fmaxf(mx, __shfl_xor(mx, 8));
      mnew[v]   = fmaxf(mi[v], mx);
      pscale[v] = __expf(mi[v] - mnew[v]);
      mi[v]     = mnew[v];
    }
#pragma unroll
    for (int v = 0; v < 8; ++v) {
      float rs = 0.f;
#pragma unroll
      for (int t = 0; t < 4; ++t) {
        const float e = __expf(st[t][v] - mnew[v]);
        st[t][v] = e;
        rs += e;
      }
      rs += __shfl_xor(rs, 1);
      rs += __shfl_xor(rs, 2);
      rs += __shfl_xor(rs, 4);
      rs += __shfl_xor(rs, 8);
      li[v] = li[v] * pscale[v] + rs;
    }

    // rescale O, spill P (D layout f32) to LDS as bf16 A layout source
#pragma unroll
    for (int t = 0; t < 4; ++t) {
#pragma unroll
      for (int v = 0; v < 8; ++v) {
        o[t][v] *= pscale[v];
        Plds[wv][(v + kh * 8) * 64 + t * 16 + ln] = (bf16)st[t][v];
      }
    }
    __builtin_amdgcn_wave_barrier();

    // O += P V : contraction over 64 keys (2 A-frags), 4 hd-tiles
#pragma unroll
    for (int f = 0; f < 2; ++f) {
      const bf16* pp = &Plds[wv][ln * 64 + f * 32 + kh * 8];
      const v16bf pa = make_frag(*(const v8bf*)pp, *(const v8bf*)(pp + 16));
#pragma unroll
      for (int t = 0; t < 4; ++t) {
        const bf16* vb = vp + (size_t)(t * 16 + ln) * Sn + kb + f * 32 + kh * 8;
        const v16bf bv = make_frag(*(const v8bf*)vb, *(const v8bf*)(vb + 16));
        o[t] = wmma_bf16(pa, bv, o[t]);
      }
    }
  }

  // epilogue: normalize and write concat layout [B*S, D] (bf16 for final GEMM)
  const int b = bh >> 4, h = bh & 15;
#pragma unroll
  for (int v = 0; v < 8; ++v) {
    const float inv = 1.0f / li[v];
    const int gm = b * Sn + qbase + v + kh * 8;
#pragma unroll
    for (int t = 0; t < 4; ++t) {
      const int col = h * HDn + t * 16 + ln;
      O[(size_t)gm * Dn + col] = (bf16)(o[t][v] * inv);
    }
  }
}

// ------------------------------- launcher ----------------------------------

extern "C" void kernel_launch(void* const* d_in, const int* in_sizes, int n_in,
                              void* d_out, int out_size, void* d_ws, size_t ws_size,
                              hipStream_t stream) {
  const float* q_in = (const float*)d_in[0];
  const float* k_in = (const float*)d_in[1];
  const float* v_in = (const float*)d_in[2];
  const float* Wq = (const float*)d_in[3];
  const float* bq = (const float*)d_in[4];
  const float* Wk = (const float*)d_in[5];
  const float* bk = (const float*)d_in[6];
  const float* Wv = (const float*)d_in[7];
  const float* bv = (const float*)d_in[8];
  const float* Wo = (const float*)d_in[9];
  const float* bo = (const float*)d_in[10];
  float* out = (float*)d_out;

  // workspace carve-up (256B aligned)
  char* ws = (char*)d_ws;
  size_t off = 0;
  auto carve = [&](size_t bytes) -> void* {
    void* p = ws + off;
    off = (off + bytes + 255) & ~(size_t)255;
    return p;
  };
  const size_t actB = (size_t)Mn * Dn * sizeof(bf16);   // 8 MB
  const size_t wB   = (size_t)Dn * Dn * sizeof(bf16);   // 2 MB
  bf16* qa   = (bf16*)carve(actB);
  bf16* ka   = (bf16*)carve(actB);
  bf16* va   = (bf16*)carve(actB);
  bf16* Wqt  = (bf16*)carve(wB);
  bf16* Wkt  = (bf16*)carve(wB);
  bf16* Wvt  = (bf16*)carve(wB);
  bf16* Wot  = (bf16*)carve(wB);
  bf16* Qb   = (bf16*)carve(actB);   // [B,H,S,HD]
  bf16* Kb   = (bf16*)carve(actB);   // [B,H,S,HD]
  bf16* Vtb  = (bf16*)carve(actB);   // [B,H,HD,S]
  bf16* Obuf = (bf16*)carve(actB);   // [B*S, D]
  (void)ws_size; (void)in_sizes; (void)n_in; (void)out_size;

  const int nAct = Mn * Dn;  // 4 194 304

  // 1) precision conversion + weight transposes
  k_cvt_f32_bf16<<<1024, 256, 0, stream>>>(q_in, qa, nAct);
  k_cvt_f32_bf16<<<1024, 256, 0, stream>>>(k_in, ka, nAct);
  k_cvt_f32_bf16<<<1024, 256, 0, stream>>>(v_in, va, nAct);
  dim3 tg(Dn / 16, Dn / 16), tb(16, 16);
  k_transpose_w<<<tg, tb, 0, stream>>>(Wq, Wqt);
  k_transpose_w<<<tg, tb, 0, stream>>>(Wk, Wkt);
  k_transpose_w<<<tg, tb, 0, stream>>>(Wv, Wvt);
  k_transpose_w<<<tg, tb, 0, stream>>>(Wo, Wot);

  // 2) projections: 16 row-groups x 16 col-panels = 256 blocks of 8 waves
  const int gemmBlocks = (Mn / 256) * (Dn / 64);  // 256
  k_gemm<0><<<gemmBlocks, 256, 0, stream>>>(qa, Wqt, bq, Qb,  nullptr, 0.125f);
  k_gemm<0><<<gemmBlocks, 256, 0, stream>>>(ka, Wkt, bk, Kb,  nullptr, 1.0f);
  k_gemm<1><<<gemmBlocks, 256, 0, stream>>>(va, Wvt, bv, Vtb, nullptr, 1.0f);

  // 3) causal flash attention: B*H*(S/64) = 1024 blocks, 4 waves each
  k_attn<<<Bn * Hn * (Sn / 64), 128, 0, stream>>>(Qb, Kb, Vtb, Obuf);

  // 4) output projection -> f32 d_out
  k_gemm<2><<<gemmBlocks, 256, 0, stream>>>(Obuf, Wot, bo, nullptr, out, 1.0f);
}
